// Generator_72945724555920
// MI455X (gfx1250) — matrix operand order repaired and checked by reading, not compiled, not run on hardware
//
#include <hip/hip_runtime.h>
#include <math.h>

// ---------------- types for CDNA5 WMMA ----------------
typedef _Float16 f16;
typedef __attribute__((ext_vector_type(16))) _Float16 v16h;
typedef __attribute__((ext_vector_type(8)))  _Float16 v8h;
typedef __attribute__((ext_vector_type(8)))  float    v8f;

#define DEV static __device__ __forceinline__

// Model constants
#define NB4   4
#define CCH   512
#define HCH   1536
#define LF    128
#define NHARM 64
#define NFFT  3840
#define HOP   960
#define NBINS 1921      // NFFT/2 + 1
#define TFR   129       // Lf + 1 frames
#define LW    122880    // LF * 960

DEV float gelu_tanh(float x) {
  float x3 = x * x * x;
  return 0.5f * x * (1.f + tanhf(0.79788456f * (x + 0.044715f * x3)));
}

// ---------------- f32 -> f16 padded weight copy: Wp[Mp,Kp] -----------------
__global__ void k_cvt_w(const float* __restrict__ W, f16* __restrict__ Wp,
                        int M, int K, int Mp, int Kp)
{
  int i = blockIdx.x * 256 + threadIdx.x;
  int total = Mp * Kp;
  if (i >= total) return;
  int k = i % Kp, m = i / Kp;
  Wp[i] = (m < M && k < K) ? (f16)W[(size_t)m * K + k] : (f16)0.f;
}

// ---- f32 [n,k,l] -> f16 transposed+padded Xt[j=(n*L+l)][k], j<Jp,k<Kp ------
__global__ void k_cvt_x(const float* __restrict__ X, f16* __restrict__ Xt,
                        int K, int L, int NB, int Kp, int Jp)
{
  int i = blockIdx.x * 256 + threadIdx.x;
  int total = Jp * Kp;
  if (i >= total) return;
  int k = i % Kp, j = i / Kp;
  int J = NB * L;
  f16 v = (f16)0.f;
  if (j < J && k < K) {
    int n = j / L, l = j - n * L;
    v = (f16)X[((size_t)n * K + k) * L + l];
  }
  Xt[i] = v;
}

// load a 16x32 f16 A/B-operand half for one lane: two b128 chunks combined
DEV v16h ld_op(const f16* p, int k0, int hl) {
  v8h lo = *(const v8h*)(p + k0 + hl * 8);
  v8h hi = *(const v8h*)(p + k0 + 16 + hl * 8);
  return __builtin_shufflevector(lo, hi, 0, 1, 2, 3, 4, 5, 6, 7,
                                 8, 9, 10, 11, 12, 13, 14, 15);
}

// =====================================================================
// WMMA GEMM on padded f16 operands. Y[n,m,l] = act(sum_k Wp[m,k]*Xt[j,k]+bias[m])
// Block = 256 threads = 8 waves (4x2); each wave owns a 32x32 output tile
// (2x2 register blocking): per 32-K step, 8 b128 loads feed 4 WMMAs.
// Block tile 128x64.  act: 0=none, 1=gelu(tanh), 2=exp
// =====================================================================
__global__ void k_gemm_wmma(const f16* __restrict__ Wp, const f16* __restrict__ Xt,
                            const float* __restrict__ bias, float* __restrict__ Y,
                            int M, int Kp, int L, int NB, int act)
{
  const int J = NB * L;
  int lane = threadIdx.x & 31;
  int wv   = threadIdx.x >> 5;
  int hl   = lane >> 4;        // half-wave: 0 or 1
  int idx  = lane & 15;
  int m0 = blockIdx.y * 128 + (wv >> 1) * 32;
  int j0 = blockIdx.x * 64 + (wv & 1) * 32;

  const f16* wr0 = Wp + (size_t)(m0 + idx) * Kp;        // A rows (padded, valid)
  const f16* wr1 = wr0 + (size_t)16 * Kp;
  const f16* xr0 = Xt + (size_t)(j0 + idx) * Kp;        // B cols (padded, valid)
  const f16* xr1 = xr0 + (size_t)16 * Kp;

  v8f acc00 = {}, acc01 = {}, acc10 = {}, acc11 = {};
  for (int k0 = 0; k0 < Kp; k0 += 32) {
    // A 16x32 f16 layout: half 0 -> K {0..7,16..23}, half 1 -> {8..15,24..31}
    v16h a0 = ld_op(wr0, k0, hl);
    v16h a1 = ld_op(wr1, k0, hl);
    // B 32x16 f16 layout: half 0 -> K {0..7,16..23}, half 1 -> {8..15,24..31}
    v16h b0 = ld_op(xr0, k0, hl);
    v16h b1 = ld_op(xr1, k0, hl);
    acc00 = __builtin_amdgcn_wmma_f32_16x16x32_f16(false, a0, false, b0, (short)0,
                                                   acc00, false, false);
    acc01 = __builtin_amdgcn_wmma_f32_16x16x32_f16(false, a0, false, b1, (short)0,
                                                   acc01, false, false);
    acc10 = __builtin_amdgcn_wmma_f32_16x16x32_f16(false, a1, false, b0, (short)0,
                                                   acc10, false, false);
    acc11 = __builtin_amdgcn_wmma_f32_16x16x32_f16(false, a1, false, b1, (short)0,
                                                   acc11, false, false);
  }

  v8f accs[2][2] = {{acc00, acc01}, {acc10, acc11}};
#pragma unroll
  for (int ji = 0; ji < 2; ++ji) {
    int jc = j0 + ji * 16 + idx;
    if (jc >= J) continue;
    int n = jc / L, l = jc - n * L;
#pragma unroll
    for (int mi = 0; mi < 2; ++mi) {
#pragma unroll
      for (int r = 0; r < 8; ++r) {
        int m = m0 + mi * 16 + hl * 8 + r;   // C/D layout: lanes16-31 -> M=8..15
        if (m >= M) continue;
        float v = accs[mi][ji][r] + bias[m];
        if (act == 1)      v = gelu_tanh(v);
        else if (act == 2) v = expf(v);
        Y[((size_t)n * M + m) * L + l] = v;
      }
    }
  }
}

// ---------------- elementwise helpers ----------------
__global__ void k_zero(float* p, int n) {
  int i = blockIdx.x * 256 + threadIdx.x; if (i < n) p[i] = 0.f;
}
__global__ void k_add_inplace(float* y, const float* h, int n) {
  int i = blockIdx.x * 256 + threadIdx.x; if (i < n) y[i] += h[i];
}
__global__ void k_copy(float* dst, const float* src, int n) {
  int i = blockIdx.x * 256 + threadIdx.x; if (i < n) dst[i] = src[i];
}

// ---------------- depthwise causal conv k=7 ----------------
__global__ void k_dwconv7(const float* __restrict__ x, const float* __restrict__ w,
                          const float* __restrict__ b, float* __restrict__ y,
                          int NB, int C, int L)
{
  int i = blockIdx.x * 256 + threadIdx.x;
  int total = NB * C * L;
  if (i >= total) return;
  int l = i % L;
  int c = (i / L) % C;
  const float* xr = x + (i - l);
  float acc = b[c];
#pragma unroll
  for (int t = 0; t < 7; ++t) {
    int s = l + t - 6;
    if (s >= 0) acc += w[c * 7 + t] * xr[s];
  }
  y[i] = acc;
}

// ---------------- channel norm (mean/var over C per position) ----------------
__global__ void k_chnorm(const float* __restrict__ x, const float* __restrict__ g,
                         const float* __restrict__ b, float* __restrict__ y,
                         int C, int L)
{
  int j = blockIdx.x;               // j = n*L + l
  int n = j / L, l = j - n * L;
  const float* col = x + ((size_t)n * C) * L + l;
  __shared__ float s1[256], s2[256];
  int tid = threadIdx.x;
  float a = 0.f, q = 0.f;
  for (int c = tid; c < C; c += 256) { float v = col[(size_t)c * L]; a += v; q += v * v; }
  s1[tid] = a; s2[tid] = q; __syncthreads();
  for (int off = 128; off; off >>= 1) {
    if (tid < off) { s1[tid] += s1[tid + off]; s2[tid] += s2[tid + off]; }
    __syncthreads();
  }
  float mu  = s1[0] / C;
  float var = s2[0] / C - mu * mu;
  float inv = rsqrtf(var + 1e-6f);
  for (int c = tid; c < C; c += 256) {
    float v = col[(size_t)c * L];
    y[((size_t)n * C + c) * L + l] = (v - mu) * inv * g[c] + b[c];
  }
}

// ---------------- harmonic oscillator: fused interp + cumsum + sin + mean ----
__global__ void k_harm(const float* __restrict__ f0, const float* __restrict__ amps,
                       float* __restrict__ harm)
{
  int h = blockIdx.x, n = blockIdx.y;
  __shared__ float sbuf[256];
  int tid = threadIdx.x;
  const float mul = (float)(h + 1);
  float carry = 0.f;
  for (int base = 0; base < LW; base += 256) {
    int p = base + tid;
    float pos = (p + 0.5f) * ((float)LF / (float)LW) - 0.5f;
    pos = fminf(fmaxf(pos, 0.f), (float)(LF - 1));
    int i0 = (int)floorf(pos);
    int i1 = min(i0 + 1, LF - 1);
    float t = pos - (float)i0;
    float fa = fmaxf(f0[n * LF + i0], 20.f);
    float fb = fmaxf(f0[n * LF + i1], 20.f);
    float freq = (fa * (1.f - t) + fb * t) * mul;       // interp(max(f0,20))*mul
    const float* ar = amps + ((size_t)n * NHARM + h) * LF;
    float av = ar[i0] * (1.f - t) + ar[i1] * t;
    float inc = freq * (1.f / 48000.f);
    sbuf[tid] = inc; __syncthreads();
    for (int off = 1; off < 256; off <<= 1) {
      float v = (tid >= off) ? sbuf[tid - off] : 0.f;
      __syncthreads();
      sbuf[tid] += v;
      __syncthreads();
    }
    float dt = carry + sbuf[tid];
    float tot = sbuf[255];
    __syncthreads();
    carry += tot;
    float w = sinf(6.283185307179586f * dt) * av * (1.f / (float)NHARM);
    atomicAdd(&harm[(size_t)n * LW + p], w);
  }
}

// ---------------- reflection pad [1,0] along time ----------------
__global__ void k_reflectpad(const float* __restrict__ x, float* __restrict__ xp,
                             int NB, int C, int L)
{
  int Lp = L + 1;
  int i = blockIdx.x * 256 + threadIdx.x;
  int total = NB * C * Lp;
  if (i >= total) return;
  int lp = i % Lp;
  int cn = i / Lp;
  int l = (lp == 0) ? 1 : (lp - 1);
  xp[i] = x[(size_t)cn * L + l];
}

// ---------------- mag/phase -> complex spectrum ----------------
__global__ void k_spec(const float* __restrict__ mp, float* __restrict__ sre,
                       float* __restrict__ sim, int NB)
{
  const int per = NBINS * TFR;
  int i = blockIdx.x * 256 + threadIdx.x;
  if (i >= NB * per) return;
  int n = i / per;
  int r = i - n * per;                 // r = k*TFR + t
  size_t nb = (size_t)n * (NFFT + 2) * TFR;
  float mag = mp[nb + r];
  float ph  = mp[nb + (size_t)NBINS * TFR + r];
  mag = expf(fminf(mag, 6.f));
  sre[i] = mag * cosf(ph);
  sim[i] = mag * sinf(ph);
}

// ---------------- twiddle tables ----------------
__global__ void k_twiddle(float* ct, float* st)
{
  int m = blockIdx.x * 256 + threadIdx.x;
  if (m < NFFT) {
    float a = 6.283185307179586f * (float)m / (float)NFFT;
    ct[m] = cosf(a);
    st[m] = sinf(a);
  }
}

// ---------------- irfft via LDS twiddle tables + incremental index ----------
__global__ void k_irfft(const float* __restrict__ sre, const float* __restrict__ sim,
                        const float* __restrict__ ct, const float* __restrict__ st,
                        float* __restrict__ frames)
{
  __shared__ float lct[NFFT], lst[NFFT];
  __shared__ float lre[NBINS], lim[NBINS];
  int nt = blockIdx.y;
  int n = nt / TFR, t = nt - n * TFR;
  for (int m = threadIdx.x; m < NFFT; m += 256) { lct[m] = ct[m]; lst[m] = st[m]; }
  for (int k = threadIdx.x; k < NBINS; k += 256) {
    size_t src = ((size_t)n * NBINS + k) * TFR + t;
    lre[k] = sre[src]; lim[k] = sim[src];
  }
  __syncthreads();
  int j = blockIdx.x * 256 + threadIdx.x;
  float acc = lre[0] + ((j & 1) ? -lre[NFFT / 2] : lre[NFFT / 2]);
  int m = 0;
  for (int k = 1; k < NFFT / 2; ++k) {
    m += j; if (m >= NFFT) m -= NFFT;        // m = (k*j) mod NFFT
    acc += 2.f * (lre[k] * lct[m] - lim[k] * lst[m]);
  }
  frames[(size_t)nt * NFFT + j] = acc * (1.f / (float)NFFT);
}

// ---------------- istft overlap-add with ones window ----------------
__global__ void k_istft(const float* __restrict__ frames, float* __restrict__ noise, int NB)
{
  int i = blockIdx.x * 256 + threadIdx.x;
  if (i >= NB * LW) return;
  int n = i / LW;
  int p = i - n * LW;
  int g = p + NFFT / 2;
  int thi = g / HOP;
  float sum = 0.f; int cnt = 0;
  int tlo = thi - 3; if (tlo < 0) tlo = 0;
  int tmax = thi; if (tmax > TFR - 1) tmax = TFR - 1;
  for (int t = tlo; t <= tmax; ++t) {
    int off = g - t * HOP;
    if (off < NFFT) { sum += frames[((size_t)(n * TFR + t)) * NFFT + off]; ++cnt; }
  }
  noise[i] = sum / (float)cnt;
}

// ---------------- combine harm+noise -> wave_raw, wave_f, d_out(raw) ----------
__global__ void k_combine(const float* __restrict__ harm, const float* __restrict__ noise,
                          float* __restrict__ wave_raw, float* __restrict__ wave_f,
                          float* __restrict__ out_raw, int n)
{
  int i = blockIdx.x * 256 + threadIdx.x;
  if (i >= n) return;
  float v = harm[i] + noise[i];
  wave_raw[i] = v; wave_f[i] = v; out_raw[i] = v;
}

// ---------------- post filter: window conv (1 -> c channels, stride s, k=2s) --
__global__ void k_winconv(const float* __restrict__ wave, const float* __restrict__ w,
                          const float* __restrict__ b, float* __restrict__ y,
                          int NB, int c, int s, int Lo)
{
  int i = blockIdx.x * 256 + threadIdx.x;
  int total = NB * c * Lo;
  if (i >= total) return;
  int q  = i % Lo;
  int ch = (i / Lo) % c;
  int n  = i / (Lo * c);
  float acc = b[ch];
  int base = q * s - s / 2;
  for (int k = 0; k < 2 * s; ++k) {
    int p = base + k;
    if (p >= 0 && p < LW) acc += w[ch * 2 * s + k] * wave[(size_t)n * LW + p];
  }
  y[i] = acc;
}

// ---------------- gelu(w) * interp(xf, Lo) ----------------
__global__ void k_pfact(const float* __restrict__ wbuf, const float* __restrict__ xf,
                        float* __restrict__ out, int NB, int c, int Lo)
{
  int i = blockIdx.x * 256 + threadIdx.x;
  int total = NB * c * Lo;
  if (i >= total) return;
  int q  = i % Lo;
  int ch = (i / Lo) % c;
  int n  = i / (Lo * c);
  float pos = (q + 0.5f) * ((float)LF / (float)Lo) - 0.5f;
  pos = fminf(fmaxf(pos, 0.f), (float)(LF - 1));
  int i0 = (int)floorf(pos);
  int i1 = min(i0 + 1, LF - 1);
  float t = pos - (float)i0;
  const float* xr = xf + ((size_t)n * c + ch) * LF;
  float xv = xr[i0] * (1.f - t) + xr[i1] * t;
  out[i] = gelu_tanh(wbuf[i]) * xv;
}

// ---------------- causal dilated conv c x c x 5 ----------------
__global__ void k_pfconv(const float* __restrict__ a, const float* __restrict__ w,
                         const float* __restrict__ b, float* __restrict__ y,
                         int NB, int c, int Lo, int d)
{
  int i = blockIdx.x * 256 + threadIdx.x;
  int total = NB * c * Lo;
  if (i >= total) return;
  int q  = i % Lo;
  int co = (i / Lo) % c;
  int n  = i / (Lo * c);
  float acc = b[co];
  for (int ci = 0; ci < c; ++ci) {
    const float* ar = a + ((size_t)n * c + ci) * Lo;
    const float* wr = w + (co * c + ci) * 5;
#pragma unroll
    for (int t = 0; t < 5; ++t) {
      int s = q + (t - 4) * d;
      if (s >= 0) acc += wr[t] * ar[s];
    }
  }
  y[i] = acc;
}

// ---------------- transpose conv (c -> 1, stride s, k=2s), accumulate --------
__global__ void k_deconv_acc(const float* __restrict__ wbuf, const float* __restrict__ w,
                             const float* __restrict__ b, float* __restrict__ wave_f,
                             int NB, int c, int s, int Lo)
{
  int i = blockIdx.x * 256 + threadIdx.x;
  if (i >= NB * LW) return;
  int p = i % LW;
  int n = i / LW;
  float acc = b[0];
  int qmax = (p + s / 2) / s;
  for (int q = qmax - 1; q <= qmax; ++q) {
    if (q < 0 || q >= Lo) continue;
    int k = p + s / 2 - q * s;
    if (k < 0 || k >= 2 * s) continue;
    for (int co = 0; co < c; ++co)
      acc += w[co * 2 * s + k] * wbuf[((size_t)n * c + co) * Lo + q];
  }
  wave_f[i] += acc;
}

// =====================================================================
// Host side.  Input leaf order: setup_inputs() dict insertion order, params
// flattened recursively in insertion order:
//   0:x 1:f0 2:fe_in_w 3:fe_in_b 4:fe_out_w 5:fe_out_b 6:amp_w 7:amp_b
//   8:mp_w 9:mp_b
//   10+8*i (i=0..7): dw_w, dw_b, ln_g, ln_b, pw1_w, pw1_b, pw2_w, pw2_b
//   74+14*u (u=0..3): feat_w, feat_b, win_w, win_b, [mid w,b]x4, wout_w, wout_b
// =====================================================================
extern "C" void kernel_launch(void* const* d_in, const int* in_sizes, int n_in,
                              void* d_out, int out_size, void* d_ws, size_t ws_size,
                              hipStream_t stream)
{
  (void)in_sizes; (void)n_in; (void)out_size; (void)ws_size;
  const float* X_IN = (const float*)d_in[0];
  const float* F0   = (const float*)d_in[1];
  const float* FE_IN_W  = (const float*)d_in[2];
  const float* FE_IN_B  = (const float*)d_in[3];
  const float* FE_OUT_W = (const float*)d_in[4];
  const float* FE_OUT_B = (const float*)d_in[5];
  const float* AMP_W    = (const float*)d_in[6];
  const float* AMP_B    = (const float*)d_in[7];
  const float* MP_W     = (const float*)d_in[8];
  const float* MP_B     = (const float*)d_in[9];

  float* out = (float*)d_out;
  float* out_f   = out;                       // wave_f
  float* out_raw = out + (size_t)NB4 * LW;    // wave_raw

  // ---- workspace carve (32B-aligned slices) ----
  char* wsb = (char*)d_ws;
  size_t off = 0;
  auto allocB = [&](size_t bytes) {
    void* p = wsb + off;
    off = (off + bytes + 31) & ~(size_t)31;
    return p;
  };
  auto allocF = [&](size_t nfl) { return (float*)allocB(nfl * 4); };
  auto allocH = [&](size_t nh)  { return (f16*)allocB(nh * 2); };

  float* feat_y     = allocF((size_t)NB4 * CCH * LF);
  float* bufA       = allocF((size_t)NB4 * CCH * LF);
  float* bufB       = allocF((size_t)NB4 * CCH * LF);
  float* bufH       = allocF((size_t)NB4 * HCH * LF);
  float* feat_final = allocF((size_t)NB4 * CCH * LF);
  float* amps       = allocF((size_t)NB4 * NHARM * LF);
  float* harm       = allocF((size_t)NB4 * LW);
  float* xp         = allocF((size_t)NB4 * CCH * TFR);
  float* mp_out     = allocF((size_t)NB4 * (NFFT + 2) * TFR);
  float* sre        = allocF((size_t)NB4 * NBINS * TFR);
  float* sim        = allocF((size_t)NB4 * NBINS * TFR);
  float* ctab       = allocF(NFFT);
  float* stab       = allocF(NFFT);
  float* frames     = allocF((size_t)NB4 * TFR * NFFT);
  float* noise      = allocF((size_t)NB4 * LW);
  float* wave_raw   = allocF((size_t)NB4 * LW);
  float* wave_f     = allocF((size_t)NB4 * LW);
  float* xf         = allocF((size_t)NB4 * 64 * LF);
  float* wA         = allocF((size_t)NB4 * 16 * (LW / 4));   // max c*Lo product
  float* wB         = allocF((size_t)NB4 * 16 * (LW / 4));
  f16*   Wp16       = allocH((size_t)3968 * 512);            // max Mp*Kp (Mp%128==0)
  f16*   Xt16       = allocH((size_t)576 * 1536);            // max Jp*Kp (Jp%64==0)

  auto ew = [&](long long n) { return dim3((unsigned)((n + 255) / 256)); };

  // padded-f16 WMMA GEMM launcher: converts W and X, then runs the GEMM
  auto gemm = [&](const float* W, const float* Xsrc, const float* B, float* Y,
                  int M, int K, int L, int act) {
    int Kp = (K + 31) & ~31;
    int Mp = (M + 127) & ~127;
    int J  = NB4 * L;
    int Jp = (J + 63) & ~63;
    k_cvt_w<<<ew((long long)Mp * Kp), 256, 0, stream>>>(W, Wp16, M, K, Mp, Kp);
    k_cvt_x<<<ew((long long)Jp * Kp), 256, 0, stream>>>(Xsrc, Xt16, K, L, NB4, Kp, Jp);
    dim3 g(Jp / 64, Mp / 128);
    k_gemm_wmma<<<g, 256, 0, stream>>>(Wp16, Xt16, B, Y, M, Kp, L, NB4, act);
  };

  // ---------- feature extractor ----------
  gemm(FE_IN_W, X_IN, FE_IN_B, feat_y, CCH, 80, LF, 0);
  for (int blk = 0; blk < 8; ++blk) {
    int base = 10 + blk * 8;
    const float* dw_w  = (const float*)d_in[base + 0];
    const float* dw_b  = (const float*)d_in[base + 1];
    const float* ln_g  = (const float*)d_in[base + 2];
    const float* ln_b  = (const float*)d_in[base + 3];
    const float* pw1_w = (const float*)d_in[base + 4];
    const float* pw1_b = (const float*)d_in[base + 5];
    const float* pw2_w = (const float*)d_in[base + 6];
    const float* pw2_b = (const float*)d_in[base + 7];
    int nelem = NB4 * CCH * LF;
    k_dwconv7<<<ew(nelem), 256, 0, stream>>>(feat_y, dw_w, dw_b, bufA, NB4, CCH, LF);
    k_chnorm<<<NB4 * LF, 256, 0, stream>>>(bufA, ln_g, ln_b, bufB, CCH, LF);
    gemm(pw1_w, bufB, pw1_b, bufH, HCH, CCH, LF, 1);   // + gelu
    gemm(pw2_w, bufH, pw2_b, bufA, CCH, HCH, LF, 0);
    k_add_inplace<<<ew(nelem), 256, 0, stream>>>(feat_y, bufA, nelem);
  }
  gemm(FE_OUT_W, feat_y, FE_OUT_B, feat_final, CCH, CCH, LF, 0);

  // ---------- harmonic oscillator ----------
  gemm(AMP_W, feat_final, AMP_B, amps, NHARM, CCH, LF, 2);   // exp fused
  k_zero<<<ew(NB4 * LW), 256, 0, stream>>>(harm, NB4 * LW);
  k_harm<<<dim3(NHARM, NB4), 256, 0, stream>>>(F0, amps, harm);

  // ---------- noise generator ----------
  k_reflectpad<<<ew(NB4 * CCH * TFR), 256, 0, stream>>>(feat_final, xp, NB4, CCH, LF);
  gemm(MP_W, xp, MP_B, mp_out, NFFT + 2, CCH, TFR, 0);
  k_spec<<<ew((long long)NB4 * NBINS * TFR), 256, 0, stream>>>(mp_out, sre, sim, NB4);
  k_twiddle<<<ew(NFFT), 256, 0, stream>>>(ctab, stab);
  k_irfft<<<dim3(NFFT / 256, NB4 * TFR), 256, 0, stream>>>(sre, sim, ctab, stab, frames);
  k_istft<<<ew(NB4 * LW), 256, 0, stream>>>(frames, noise, NB4);

  // ---------- combine ----------
  k_combine<<<ew(NB4 * LW), 256, 0, stream>>>(harm, noise, wave_raw, wave_f, out_raw,
                                              NB4 * LW);

  // ---------- post filter ----------
  const int PF_S[4] = {4, 12, 96, 128};
  const int PF_C[4] = {16, 32, 64, 64};
  for (int u = 0; u < 4; ++u) {
    int base = 74 + u * 14;
    const float* feat_w = (const float*)d_in[base + 0];
    const float* feat_b = (const float*)d_in[base + 1];
    const float* win_w  = (const float*)d_in[base + 2];
    const float* win_b  = (const float*)d_in[base + 3];
    const float* wout_w = (const float*)d_in[base + 12];
    const float* wout_b = (const float*)d_in[base + 13];
    int s = PF_S[u], c = PF_C[u], Lo = LW / s;
    int nelem = NB4 * c * Lo;
    gemm(feat_w, feat_final, feat_b, xf, c, CCH, LF, 0);
    k_winconv<<<ew(nelem), 256, 0, stream>>>(wave_raw, win_w, win_b, wA, NB4, c, s, Lo);
    for (int i = 0; i < 4; ++i) {
      const float* mw = (const float*)d_in[base + 4 + 2 * i];
      const float* mb = (const float*)d_in[base + 5 + 2 * i];
      k_pfact<<<ew(nelem), 256, 0, stream>>>(wA, xf, wB, NB4, c, Lo);
      k_pfconv<<<ew(nelem), 256, 0, stream>>>(wB, mw, mb, wA, NB4, c, Lo, 1 << i);
    }
    k_deconv_acc<<<ew(NB4 * LW), 256, 0, stream>>>(wA, wout_w, wout_b, wave_f,
                                                   NB4, c, s, Lo);
  }

  // ---------- final output ----------
  k_copy<<<ew(NB4 * LW), 256, 0, stream>>>(out_f, wave_f, NB4 * LW);
}
